// CIM_layer_13185549599074
// MI455X (gfx1250) — compile-verified, matrix operand order checked by reading, not run on hardware
//
#include <hip/hip_runtime.h>
#include <hip/hip_bf16.h>

// ---------------- problem constants ----------------
#define NN        4096
#define NC        20        // C
#define NCOLS     21        // C+1
#define KEEP      410       // ceil(0.1 * 4096)
#define CON_THR   0.85f
#define NMS_THR   0.25f
#define CLS_THR   0.25f
#define IOU_THR   0.5f
#define ROWCNT_LIM 3686     // count < 0.9*N (=3686.4)  <=>  count <= 3686

typedef int v8i __attribute__((ext_vector_type(8)));

// ---------------- workspace layout (bytes) ----------------
#define OFF_ROWFLAG 0          // int[4096]          16384
#define OFF_GTLAB   16384      // int[4096]          16384
#define OFF_GTW     32768      // float[4096]        16384
#define OFF_NMS     49152      // u8[20*4096]        81920
#define OFF_CAND    131072     // u8[20*4096]        81920
#define OFF_SUBIOU  212992     // float[20*410*410]  ~13.45 MB

// =====================================================================
// Kernel 0: zero the candidate mask (20*4096 bytes = 20480 words)
// =====================================================================
__global__ void k_zero_cand(unsigned int* p) {
    p[blockIdx.x * 256 + threadIdx.x] = 0u;
}

// =====================================================================
// Kernel A: rowflag[i] = (sum_j (asy[i,j] > CON_THR) <= 3686)
// One wave per 16-row tile. The 0/1 mask row-sum is computed as
// mask(16x64 u8) x ones(64x16 u8) on V_WMMA_I32_16X16X64_IU8,
// accumulating i32 across the K loop (load-bound; XDL reduction is free).
// =====================================================================
__global__ void k_asy_rowcount(const float* __restrict__ asy,
                               int* __restrict__ rowflag) {
    const int tile = blockIdx.x;            // 0..255 -> rows tile*16 .. +15
    const int lane = threadIdx.x;           // 0..31 (one wave)
    const int r    = lane & 15;             // A-matrix: lanes L and L+16 hold row M = L&15
    const int koff = (lane < 16) ? 0 : 8;   // K-half split per 8-bit A layout
    const float* rowp = asy + (size_t)(tile * 16 + r) * NN;

    v8i acc = {0, 0, 0, 0, 0, 0, 0, 0};
    // B = ones: every u8 byte == 1
    const v8i bones = {0x01010101, 0x01010101, 0x01010101, 0x01010101,
                       0x01010101, 0x01010101, 0x01010101, 0x01010101};

    for (int k0 = 0; k0 < NN; k0 += 64) {
        v8i a;
#pragma unroll
        for (int v = 0; v < 8; ++v) {
            // 8-bit A 16x64 layout: VGPR v holds K = (v>>1)*16 + (v&1)*4 + koff .. +3
            const int kk = k0 + ((v >> 1) << 4) + ((v & 1) << 2) + koff;
            const float4 f = *reinterpret_cast<const float4*>(rowp + kk);
            unsigned d = (f.x > CON_THR ? 1u : 0u)
                       | ((f.y > CON_THR ? 1u : 0u) << 8)
                       | ((f.z > CON_THR ? 1u : 0u) << 16)
                       | ((f.w > CON_THR ? 1u : 0u) << 24);
            a[v] = (int)d;
        }
        // D[m][n] = sum_k A[m][k] * 1  (all n columns equal the row count)
        acc = __builtin_amdgcn_wmma_i32_16x16x64_iu8(
            /*sgn_a=*/false, a, /*sgn_b=*/false, bones, acc,
            /*reuse_a=*/false, /*reuse_b=*/false);
    }

    // 16x16 i32 C/D layout: VGPR p -> M=p (lanes 0-15) / M=p+8 (lanes 16-31)
    if (lane == 0) {
#pragma unroll
        for (int m = 0; m < 8; ++m)
            rowflag[tile * 16 + m] = (acc[m] <= ROWCNT_LIM) ? 1 : 0;
    }
    if (lane == 16) {
#pragma unroll
        for (int m = 0; m < 8; ++m)
            rowflag[tile * 16 + 8 + m] = (acc[m] <= ROWCNT_LIM) ? 1 : 0;
    }
}

// =====================================================================
// Kernel B: per-class (one 1024-thread block per class):
//   1) stable descending sort of cls[:,c] via bitonic on (value, index)
//   2) gather 410x410 sub-IoU into workspace (full-block parallel)
//   3) greedy NMS, double-buffered LDS row pipeline (410 steps)
//   4) write nms_full[c] byte mask
// =====================================================================
__global__ void k_sort_nms(const float* __restrict__ pcls,
                           const float* __restrict__ iou,
                           unsigned char* __restrict__ nms,
                           float* __restrict__ subiou_all) {
    const int c   = blockIdx.x;     // class 0..19
    const int tid = threadIdx.x;    // 0..1023

    __shared__ float sv[NN];
    __shared__ int   si[NN];
    __shared__ int   sup[KEEP];
    __shared__ float rowbuf[2][KEEP];

    // load cls column c (cls = predict_cls[:,1:], so column c+1)
    for (int t = tid; t < NN; t += 1024) {
        sv[t] = pcls[(size_t)t * NCOLS + c + 1];
        si[t] = t;
    }
    __syncthreads();

    // Bitonic sort with total order: value desc, index asc on ties
    // (== stable argsort of -cls). "pre(a,b)" = a precedes b.
    for (int k = 2; k <= NN; k <<= 1) {
        for (int j = k >> 1; j > 0; j >>= 1) {
            for (int t = tid; t < NN; t += 1024) {
                const int ixj = t ^ j;
                if (ixj > t) {
                    const bool up = ((t & k) == 0);
                    const float v0 = sv[t], v1 = sv[ixj];
                    const int   i0 = si[t], i1 = si[ixj];
                    // does element at ixj precede element at t?
                    const bool pre = (v1 > v0) || (v1 == v0 && i1 < i0);
                    if (up == pre) {
                        sv[t] = v1; sv[ixj] = v0;
                        si[t] = i1; si[ixj] = i0;
                    }
                }
            }
            __syncthreads();
        }
    }

    // Pre-gather sub-IoU: subiou[i*KEEP+t] = iou[si[i], si[t]]
    float* sub = subiou_all + (size_t)c * KEEP * KEEP;
    for (int e = tid; e < KEEP * KEEP; e += 1024) {
        const int ii = e / KEEP;
        const int tt = e - ii * KEEP;
        sub[e] = iou[(size_t)si[ii] * NN + si[tt]];
    }
    for (int t = tid; t < KEEP; t += 1024) sup[t] = 0;
    __threadfence_block();
    __syncthreads();

    // prime row 0
    for (int t = tid; t < KEEP; t += 1024) rowbuf[0][t] = sub[t];
    __syncthreads();

    // Greedy NMS: item i (if kept) suppresses later j with iou >= thr
    for (int i = 0; i < KEEP; ++i) {
        const int cb = i & 1, nb = (i + 1) & 1;
        // prefetch next row unconditionally (hides L2 latency behind this step)
        if (i + 1 < KEEP) {
            for (int t = tid; t < KEEP; t += 1024)
                rowbuf[nb][t] = sub[(size_t)(i + 1) * KEEP + t];
        }
        if (!sup[i]) {
            for (int t = i + 1 + tid; t < KEEP; t += 1024)
                if (rowbuf[cb][t] >= NMS_THR) sup[t] = 1;
        }
        __syncthreads();
    }

    // nms_full: byte mask over all N boxes
    unsigned char* nmsc = nms + (size_t)c * NN;
    for (int t = tid; t < NN; t += 1024) nmsc[t] = 0;
    __syncthreads();
    for (int t = tid; t < KEEP; t += 1024)
        if (!sup[t]) nmsc[si[t]] = 1;
}

// =====================================================================
// Kernel C: one pass over asy_iou_map serving ALL 20 classes:
// per column j, per class c: first-argmax_i of det[i,c] where
// col_base[i,j] = (asy[i,j] > CON_THR) && rowflag[i]; then scatter
// cand[c][best_i] = 1 if found && nms_full[c][j].
// Thread = column j (coalesced row reads); det rows staged in LDS.
// =====================================================================
__global__ void k_col_argmax(const float* __restrict__ asy,
                             const float* __restrict__ pdet,
                             const int* __restrict__ rowflag,
                             const unsigned char* __restrict__ nms,
                             unsigned char* __restrict__ cand) {
    const int j = blockIdx.x * 256 + threadIdx.x;   // 16 blocks x 256

    __shared__ float sdet[64][NC];
    __shared__ int   sflag[64];

    float best[NC];
    int   bidx[NC];
#pragma unroll
    for (int c = 0; c < NC; ++c) { best[c] = 0.0f; bidx[c] = -1; }

    for (int base = 0; base < NN; base += 64) {
        // stage det rows [base, base+64) (det = predict_det[:,1:])
        for (int t = threadIdx.x; t < 64 * NC; t += 256) {
            const int r = t / NC, cc = t - r * NC;
            sdet[r][cc] = pdet[(size_t)(base + r) * NCOLS + cc + 1];
        }
        for (int t = threadIdx.x; t < 64; t += 256)
            sflag[t] = rowflag[base + t];
        __syncthreads();

        if (base + 64 < NN)
            __builtin_prefetch(&asy[(size_t)(base + 64) * NN + j], 0, 0);

        for (int r = 0; r < 64; ++r) {
            const float a = asy[(size_t)(base + r) * NN + j];
            if (a > CON_THR && sflag[r]) {
                const int i = base + r;
#pragma unroll
                for (int c = 0; c < NC; ++c) {
                    const float d = sdet[r][c];
                    if (d > best[c]) { best[c] = d; bidx[c] = i; }  // strict > => first argmax
                }
            }
        }
        __syncthreads();
    }

#pragma unroll
    for (int c = 0; c < NC; ++c) {
        if (bidx[c] >= 0 && nms[(size_t)c * NN + j])
            cand[(size_t)c * NN + bidx[c]] = 1;   // benign same-value race
    }
}

// =====================================================================
// Kernel D: sequential per-class gt update (order matters: c = 0..19)
//   upd = cand & (preds_c > gt_weights) & active
// =====================================================================
__global__ void k_select_gt(const float* __restrict__ pcls,
                            const float* __restrict__ pdet,
                            const int* __restrict__ labels,
                            const unsigned char* __restrict__ cand,
                            int* __restrict__ gtlab,
                            float* __restrict__ gtw) {
    const int i = blockIdx.x * 256 + threadIdx.x;
    float w = -1.0f;
    int lab = 0;
    for (int c = 0; c < NC; ++c) {
        if (labels[c] > 0 && cand[(size_t)c * NN + i]) {
            const float p = pcls[(size_t)i * NCOLS + c + 1] *
                            pdet[(size_t)i * NCOLS + c + 1];
            if (p > w) { w = p; lab = c + 1; }
        }
    }
    gtlab[i] = lab;
    gtw[i]   = w;
}

// =====================================================================
// Kernel E: per row i, first-argmax over j of where(gt_mask[j], iou, -1),
// then assemble the three outputs. One wave (32 lanes) per row.
// =====================================================================
__global__ void k_rowmax_out(const float* __restrict__ iou,
                             const int* __restrict__ gtlab,
                             const float* __restrict__ gtw,
                             float* __restrict__ out_labels,
                             float* __restrict__ out_iou,
                             float* __restrict__ out_lw) {
    const int warp = threadIdx.x >> 5;
    const int lane = threadIdx.x & 31;
    const int row  = blockIdx.x * 8 + warp;   // 512 blocks x 256 threads
    const float* rp = iou + (size_t)row * NN;

    float bv = -2.0f;
    int   bj = 0x7fffffff;
    for (int j = lane; j < NN; j += 32) {
        const float v = (gtlab[j] > 0) ? rp[j] : -1.0f;
        if (v > bv || (v == bv && j < bj)) { bv = v; bj = j; }
    }
    // butterfly reduce: max value, min index on ties (first occurrence)
#pragma unroll
    for (int off = 16; off > 0; off >>= 1) {
        const float ov = __shfl_xor(bv, off, 32);
        const int   oj = __shfl_xor(bj, off, 32);
        if (ov > bv || (ov == bv && oj < bj)) { bv = ov; bj = oj; }
    }

    const int   lab    = gtlab[bj];
    const float w      = gtw[bj];
    const bool  ignore = (bv == 0.0f);
    const bool  bg     = (bv < CLS_THR) && !ignore;

    if (lane < NCOLS) {
        float val;
        if (bg)          val = (lane == 0) ? 1.0f : 0.0f;
        else if (ignore) val = 0.0f;
        else             val = (lab > 0 && lane == lab) ? 1.0f : 0.0f;
        out_labels[(size_t)row * NCOLS + lane] = val;
    }
    if (lane == 0) {
        out_iou[row] = (bv > IOU_THR) ? 1.0f : 0.0f;
        out_lw[row]  = ignore ? 0.0f : w;
    }
}

// =====================================================================
extern "C" void kernel_launch(void* const* d_in, const int* in_sizes, int n_in,
                              void* d_out, int out_size, void* d_ws, size_t ws_size,
                              hipStream_t stream) {
    const float* pcls   = (const float*)d_in[0];  // predict_cls  (4096,21)
    const float* pdet   = (const float*)d_in[1];  // predict_det  (4096,21)
    // d_in[2] = rois (unused by the computation)
    const int*   labels = (const int*)d_in[3];    // (20,) int32
    const float* iou    = (const float*)d_in[4];  // (4096,4096)
    const float* asy    = (const float*)d_in[5];  // (4096,4096)

    char* ws = (char*)d_ws;
    int*           rowflag = (int*)(ws + OFF_ROWFLAG);
    int*           gtlab   = (int*)(ws + OFF_GTLAB);
    float*         gtw     = (float*)(ws + OFF_GTW);
    unsigned char* nms     = (unsigned char*)(ws + OFF_NMS);
    unsigned char* cand    = (unsigned char*)(ws + OFF_CAND);
    float*         subiou  = (float*)(ws + OFF_SUBIOU);

    float* out_labels = (float*)d_out;            // (4096,21)
    float* out_iou    = out_labels + NN * NCOLS;  // (4096,)
    float* out_lw     = out_iou + NN;             // (4096,)

    k_zero_cand   <<<80,  256,  0, stream>>>((unsigned int*)cand);   // 20480 words
    k_asy_rowcount<<<256, 32,   0, stream>>>(asy, rowflag);          // WMMA row counts
    k_sort_nms    <<<NC,  1024, 0, stream>>>(pcls, iou, nms, subiou);
    k_col_argmax  <<<16,  256,  0, stream>>>(asy, pdet, rowflag, nms, cand);
    k_select_gt   <<<16,  256,  0, stream>>>(pcls, pdet, labels, cand, gtlab, gtw);
    k_rowmax_out  <<<512, 256,  0, stream>>>(iou, gtlab, gtw, out_labels, out_iou, out_lw);
}
